// BertCosAttention_12670153523551
// MI455X (gfx1250) — matrix operand order repaired and checked by reading, not compiled, not run on hardware
//
#include <hip/hip_runtime.h>

#define S_LEN 2048
#define BATCH 8
#define HIDN  1024
#define NH    16
#define DH    64
#define BH    (BATCH*NH)   // 128

typedef __attribute__((ext_vector_type(16))) __bf16 v16bf;
typedef __attribute__((ext_vector_type(8)))  __bf16 v8bf;
typedef __attribute__((ext_vector_type(8)))  float  v8f;

// RNE float->bf16 without relying on softfloat libcalls
__device__ inline __bf16 f2bf(float f) {
    union { float f; unsigned u; } a; a.f = f;
    unsigned r = (a.u + 0x7FFFu + ((a.u >> 16) & 1u)) >> 16;
    union { unsigned short s; __bf16 b; } o; o.s = (unsigned short)r;
    return o.b;
}

// A fragment (16x32 bf16, row-major, contiguous k order, same as B):
// lane holds row M=lane&15; K = half*16 + [0..15] consecutive in elems 0..15
__device__ inline v16bf load_a(const __bf16* p) {
    union { v16bf v; v8bf h[2]; } u;
    u.h[0] = *(const v8bf*)(p);
    u.h[1] = *(const v8bf*)(p + 8);
    return u.v;
}
// B fragment (32x16 bf16) from transposed storage BT[n][k]:
// lane holds col N=lane&15; K = half*16 + [0..15] consecutive in elems 0..15
__device__ inline v16bf load_b(const __bf16* p) {
    union { v16bf v; v8bf h[2]; } u;
    u.h[0] = *(const v8bf*)(p);
    u.h[1] = *(const v8bf*)(p + 8);
    return u.v;
}

__device__ inline v8f wmma_bf16(v16bf a, v16bf b, v8f c) {
    return __builtin_amdgcn_wmma_f32_16x16x32_bf16(false, a, false, b, (short)0, c,
                                                   false, false);
}

// ---------------- support kernels ----------------

__global__ __launch_bounds__(256) void count_kernel(const int* __restrict__ mask,
                                                    float* __restrict__ counts) {
    __shared__ int sm[256];
    int b = blockIdx.x, t = threadIdx.x;
    int c = 0;
    for (int i = t; i < S_LEN; i += 256) c += (mask[b * S_LEN + i] == 0);
    sm[t] = c; __syncthreads();
    for (int s = 128; s > 0; s >>= 1) { if (t < s) sm[t] += sm[t + s]; __syncthreads(); }
    if (t == 0) counts[b] = (float)sm[0];
}

__global__ __launch_bounds__(256) void xconv_kernel(const float* __restrict__ x,
                                                    __bf16* __restrict__ xb) {
    int i = blockIdx.x * 256 + threadIdx.x;
    xb[i] = f2bf(x[i]);
}

// Wt[z][o][i] = W_z[i][o], converted to bf16 (LDS-tiled transpose)
__global__ __launch_bounds__(256) void wt_kernel(const float* __restrict__ Wq,
                                                 const float* __restrict__ Wk,
                                                 const float* __restrict__ Wv,
                                                 __bf16* __restrict__ Wt) {
    __shared__ float tile[32][33];
    const float* W = blockIdx.z == 0 ? Wq : (blockIdx.z == 1 ? Wk : Wv);
    int i0 = blockIdx.x * 32;
    int o0 = blockIdx.y * 32;
    int tx = threadIdx.x & 31, ty = threadIdx.x >> 5;
    for (int r = ty; r < 32; r += 8)
        tile[r][tx] = W[(size_t)(i0 + r) * HIDN + o0 + tx];
    __syncthreads();
    __bf16* dst = Wt + (size_t)blockIdx.z * HIDN * HIDN;
    for (int r = ty; r < 32; r += 8)
        dst[(size_t)(o0 + r) * HIDN + i0 + tx] = f2bf(tile[tx][r]);
}

// ---------------- fused QKV GEMM + mask/norm/scale epilogue ----------------
// grid: (B*S/128, NH, 3). Block computes 128 rows x one head's 64 cols.
// 8 waves, each owning a 2x2 group of 16x16 C tiles (4 WMMAs / 8 loads per k-step).
__global__ __launch_bounds__(256) void qkv_gemm(
    const __bf16* __restrict__ Xb, const __bf16* __restrict__ Wt,
    const float* __restrict__ bq, const float* __restrict__ bk,
    const float* __restrict__ bv, const int* __restrict__ mask,
    const float* __restrict__ counts,
    __bf16* __restrict__ qn, __bf16* __restrict__ knT, __bf16* __restrict__ vsT) {
    __shared__ float lds[128 * 68];
    __shared__ float rowfac[128];

    const int wave = threadIdx.x >> 5;
    const int lane = threadIdx.x & 31;
    const int lrow = lane & 15;
    const int half = lane >> 4;
    const int mp   = wave >> 1;        // m subtiles {2mp, 2mp+1} of 8
    const int np   = wave & 1;         // n subtiles {2np, 2np+1} of 4

    const int mbase = blockIdx.x * 128;
    const int head  = blockIdx.y;
    const int which = blockIdx.z;      // 0=q 1=k 2=v

    const __bf16* W    = Wt + (size_t)which * HIDN * HIDN;
    const float*  bias = which == 0 ? bq : (which == 1 ? bk : bv);

    const __bf16* ap0 = Xb + (size_t)(mbase + mp * 32 + lrow) * HIDN + half * 16;
    const __bf16* ap1 = ap0 + (size_t)16 * HIDN;
    const __bf16* bp0 = W + (size_t)(head * DH + np * 32 + lrow) * HIDN + half * 16;
    const __bf16* bp1 = bp0 + (size_t)16 * HIDN;

    v8f c00 = {}; v8f c01 = {}; v8f c10 = {}; v8f c11 = {};
    for (int k0 = 0; k0 < HIDN; k0 += 32) {
        v16bf a0 = load_a(ap0 + k0);
        v16bf a1 = load_a(ap1 + k0);
        v16bf b0 = load_b(bp0 + k0);
        v16bf b1 = load_b(bp1 + k0);
        c00 = wmma_bf16(a0, b0, c00);
        c01 = wmma_bf16(a0, b1, c01);
        c10 = wmma_bf16(a1, b0, c10);
        c11 = wmma_bf16(a1, b1, c11);
    }

    float bias0 = bias[head * DH + np * 32 + lrow];
    float bias1 = bias[head * DH + np * 32 + 16 + lrow];
#pragma unroll
    for (int g = 0; g < 8; ++g) {
        int r0 = mp * 32 + g + 8 * half;
        int r1 = r0 + 16;
        lds[r0 * 68 + np * 32 + lrow]      = c00[g] + bias0;
        lds[r0 * 68 + np * 32 + 16 + lrow] = c01[g] + bias1;
        lds[r1 * 68 + np * 32 + lrow]      = c10[g] + bias0;
        lds[r1 * 68 + np * 32 + 16 + lrow] = c11[g] + bias1;
    }
    __syncthreads();

    const int b  = mbase >> 11;          // row / 2048
    const int s0 = mbase & (S_LEN - 1);
    if (threadIdx.x < 128) {
        int r = threadIdx.x;
        int valid = (mask[b * S_LEN + s0 + r] == 0);
        float f;
        if (which == 2) {
            f = valid ? 1.0f / counts[b] : 0.0f;
        } else {
            float ss = 0.f;
            for (int d = 0; d < DH; ++d) { float x = lds[r * 68 + d]; ss += x * x; }
            f = valid ? 1.0f / fmaxf(sqrtf(ss), 1e-12f) : 0.0f;
        }
        rowfac[r] = f;
    }
    __syncthreads();

    const int bh = b * NH + head;
    if (which == 0) {
        // qn[bh][s][e], row-major (A operand of ctx GEMM)
        int r  = threadIdx.x >> 1;             // 0..127
        int dq = (threadIdx.x & 1) * 32;       // 0 or 32
        __bf16* dst = qn + (size_t)bh * S_LEN * DH + (size_t)(s0 + r) * DH + dq;
        float f = rowfac[r];
#pragma unroll
        for (int i = 0; i < 32; ++i) dst[i] = f2bf(lds[r * 68 + dq + i] * f);
    } else {
        // knT/vsT [bh][e|d][s], transposed (A / BT operands of kv GEMM)
        __bf16* base = (which == 1 ? knT : vsT) + (size_t)bh * DH * S_LEN;
        int d  = threadIdx.x >> 2;             // 0..63
        int rq = (threadIdx.x & 3) * 32;       // 0,32,64,96
        __bf16* dst = base + (size_t)d * S_LEN + s0 + rq;
#pragma unroll
        for (int i = 0; i < 32; ++i)
            dst[i] = f2bf(lds[(rq + i) * 68 + d] * rowfac[rq + i]);
    }
}

// ---------------- kv[e][d] = sum_s k[s][e] * v[s][d], per (b,h) ----------------
// grid: (BH). A = knT (64 x 2048), BT = vsT (64 x 2048). Output stored transposed kvT[d][e].
__global__ __launch_bounds__(256) void kv_gemm(const __bf16* __restrict__ knT,
                                               const __bf16* __restrict__ vsT,
                                               __bf16* __restrict__ kvT) {
    const int wave = threadIdx.x >> 5;
    const int lane = threadIdx.x & 31;
    const int lrow = lane & 15;
    const int half = lane >> 4;
    const int mi   = wave >> 1;
    const int j0   = (wave & 1) * 2;
    const int bh   = blockIdx.x;

    const __bf16* A  = knT + (size_t)bh * DH * S_LEN;
    const __bf16* Bt = vsT + (size_t)bh * DH * S_LEN;
    const __bf16* ap  = A + (size_t)(mi * 16 + lrow) * S_LEN + half * 16;
    const __bf16* bp0 = Bt + (size_t)(j0 * 16 + lrow) * S_LEN + half * 16;
    const __bf16* bp1 = bp0 + (size_t)16 * S_LEN;

    v8f c0 = {}; v8f c1 = {};
    for (int k0 = 0; k0 < S_LEN; k0 += 32) {
        v16bf a  = load_a(ap + k0);
        v16bf b0 = load_b(bp0 + k0);
        v16bf b1 = load_b(bp1 + k0);
        c0 = wmma_bf16(a, b0, c0);
        c1 = wmma_bf16(a, b1, c1);
    }

    __bf16* dst = kvT + (size_t)bh * DH * DH;
    int d0 = j0 * 16 + lrow, d1 = d0 + 16;
#pragma unroll
    for (int g = 0; g < 8; ++g) {
        int e = mi * 16 + g + 8 * half;
        dst[d0 * DH + e] = f2bf(c0[g]);   // store transposed: kvT[d][e]
        dst[d1 * DH + e] = f2bf(c1[g]);
    }
}

// ---------------- ctx[s][d] = sum_e q[s][e] * kv[e][d], per (b,h) ----------------
// grid: (S/128, BH). 128-row tiles, 2x2 subtiles per wave. Writes f32 output [B,S,H*DH].
__global__ __launch_bounds__(256) void ctx_gemm(const __bf16* __restrict__ qn,
                                                const __bf16* __restrict__ kvT,
                                                float* __restrict__ out) {
    const int wave = threadIdx.x >> 5;
    const int lane = threadIdx.x & 31;
    const int lrow = lane & 15;
    const int half = lane >> 4;
    const int mp   = wave >> 1;
    const int np   = wave & 1;
    const int sbase = blockIdx.x * 128;
    const int bh    = blockIdx.y;
    const int b = bh >> 4, head = bh & 15;

    const __bf16* A  = qn + (size_t)bh * S_LEN * DH;
    const __bf16* Bt = kvT + (size_t)bh * DH * DH;
    const __bf16* ap0 = A + (size_t)(sbase + mp * 32 + lrow) * DH + half * 16;
    const __bf16* ap1 = ap0 + (size_t)16 * DH;
    const __bf16* bp0 = Bt + (size_t)(np * 32 + lrow) * DH + half * 16;
    const __bf16* bp1 = bp0 + (size_t)16 * DH;

    v8f c00 = {}; v8f c01 = {}; v8f c10 = {}; v8f c11 = {};
#pragma unroll
    for (int k0 = 0; k0 < DH; k0 += 32) {
        v16bf a0 = load_a(ap0 + k0);
        v16bf a1 = load_a(ap1 + k0);
        v16bf b0 = load_b(bp0 + k0);
        v16bf b1 = load_b(bp1 + k0);
        c00 = wmma_bf16(a0, b0, c00);
        c01 = wmma_bf16(a0, b1, c01);
        c10 = wmma_bf16(a1, b0, c10);
        c11 = wmma_bf16(a1, b1, c11);
    }

    int d0 = head * DH + np * 32 + lrow;
#pragma unroll
    for (int g = 0; g < 8; ++g) {
        int s0r = sbase + mp * 32 + g + 8 * half;
        int s1r = s0r + 16;
        float* o0 = out + (size_t)(b * S_LEN + s0r) * HIDN;
        float* o1 = out + (size_t)(b * S_LEN + s1r) * HIDN;
        o0[d0]      = c00[g];
        o0[d0 + 16] = c01[g];
        o1[d0]      = c10[g];
        o1[d0 + 16] = c11[g];
    }
}

// ---------------- launch ----------------

extern "C" void kernel_launch(void* const* d_in, const int* in_sizes, int n_in,
                              void* d_out, int out_size, void* d_ws, size_t ws_size,
                              hipStream_t stream) {
    const float* hidden = (const float*)d_in[0];
    const int*   mask   = (const int*)d_in[1];
    const float* Wq     = (const float*)d_in[2];
    const float* bq     = (const float*)d_in[3];
    const float* Wk     = (const float*)d_in[4];
    const float* bk     = (const float*)d_in[5];
    const float* Wv     = (const float*)d_in[6];
    const float* bv     = (const float*)d_in[7];
    float* out = (float*)d_out;

    char* ws = (char*)d_ws;
    const size_t nX  = (size_t)BATCH * S_LEN * HIDN;            // 16,777,216
    const size_t oXb  = 0;
    const size_t oWt  = oXb + nX * sizeof(__bf16);              // 32 MB
    const size_t oQn  = oWt + (size_t)3 * HIDN * HIDN * sizeof(__bf16);
    const size_t oKnT = oQn + nX * sizeof(__bf16);
    const size_t oVsT = oKnT + nX * sizeof(__bf16);
    const size_t oKvT = oVsT + nX * sizeof(__bf16);
    const size_t oCnt = oKvT + (size_t)BH * DH * DH * sizeof(__bf16);

    __bf16* Xb   = (__bf16*)(ws + oXb);
    __bf16* Wt   = (__bf16*)(ws + oWt);
    __bf16* qn   = (__bf16*)(ws + oQn);
    __bf16* knT  = (__bf16*)(ws + oKnT);
    __bf16* vsT  = (__bf16*)(ws + oVsT);
    __bf16* kvT  = (__bf16*)(ws + oKvT);
    float*  cnts = (float*)(ws + oCnt);

    count_kernel<<<BATCH, 256, 0, stream>>>(mask, cnts);
    xconv_kernel<<<(unsigned)(nX / 256), 256, 0, stream>>>(hidden, Xb);
    wt_kernel<<<dim3(32, 32, 3), 256, 0, stream>>>(Wq, Wk, Wv, Wt);
    qkv_gemm<<<dim3((BATCH * S_LEN) / 128, NH, 3), 256, 0, stream>>>(
        Xb, Wt, bq, bk, bv, mask, cnts, qn, knT, vsT);
    kv_gemm<<<BH, 256, 0, stream>>>(knT, vsT, kvT);
    ctx_gemm<<<dim3(S_LEN / 128, BH), 256, 0, stream>>>(qn, kvT, out);
}
